// enc_graph_29472065585643
// MI455X (gfx1250) — matrix operand order repaired and checked by reading, not compiled, not run on hardware
//
#include <hip/hip_runtime.h>
#include <hip/hip_bf16.h>

// ---------- dims ----------
#define BB   2048
#define NN   64
#define FA   62
#define FBN  6
#define DD   5
#define CC   128
#define COUT 16
#define KK   113
#define KPADC 128                 // conv: per-channel K padded 113 -> 128
#define NCHC  (NN*KPADC/32)       // 256 conv K-chunks
#define RTOT (BB*NN)              // 131072 rows
#define EPS  1e-5f

// ---------- WMMA types ----------
typedef __attribute__((ext_vector_type(16))) __bf16 v16bf;
typedef __attribute__((ext_vector_type(8)))  float  v8f;

union Frag {
    v16bf v;
    unsigned short u[16];
    unsigned int   dw[8];
    uint4          q[2];
};

__device__ __forceinline__ unsigned short f2bf(float f) {
    union { float f; unsigned u; } x; x.f = f;
    unsigned r = x.u + 0x7FFFu + ((x.u >> 16) & 1u);   // round-to-nearest-even
    return (unsigned short)(r >> 16);
}
__device__ __forceinline__ float bf2f(unsigned short h) {
    union { unsigned u; float f; } x; x.u = ((unsigned)h) << 16;
    return x.f;
}

// ---------- ws layout ----------
#define F_SUM1 0
#define F_SQ1  128
#define F_SUM2 256
#define F_SQ2  384
#define F_SUM3 512
#define F_SQ3  640
#define F_SUM4 768
#define F_SQ4  784
#define F_SC1  1024
#define F_SH1  1152
#define F_SC2  1280
#define F_SH2  1408
#define F_SC3  1536
#define F_SH3  1664
#define F_SC4  1792
#define F_SH4  1808
#define OFF_BSUM   8192u                    // f32 [R][6]           = 3,145,728 B
#define OFF_WP1    3153920u                 // bf16 pack layer1     =   122,880 B (nch=3)
#define OFF_WP2    3276800u                 // bf16 pack layer2     =   204,800 B (nch=5)
#define OFF_WP3    3481600u                 // bf16 pack layer3     =   204,800 B
#define OFF_WC     3686400u                 // bf16 pack conv       =   262,144 B (256 chunks)
#define OFF_Z1     3948544u                 // bf16 [R][128]        = 33,554,432 B
#define OFF_Z2     37502976u                // bf16 [R][128]
#define OFF_Z4     71057408u                // f32  [B][16][16]     =  2,097,152 B

// =====================================================================
// prep kernels
// =====================================================================
__global__ void zero_stats_kernel(float* ws) {
    int i = threadIdx.x;
    if (i < 1024) ws[i] = 0.0f;
}

__global__ void bsum_kernel(const float* __restrict__ bonds, float* __restrict__ bsum) {
    int r = blockIdx.x * 256 + threadIdx.x;
    if (r >= RTOT) return;
    const float* bp = bonds + (size_t)r * (DD * FBN);
    float s[FBN];
    #pragma unroll
    for (int j = 0; j < FBN; j++) s[j] = 0.f;
    #pragma unroll
    for (int d = 0; d < DD; d++)
        #pragma unroll
        for (int j = 0; j < FBN; j++) s[j] += bp[d * FBN + j];
    #pragma unroll
    for (int j = 0; j < FBN; j++) bsum[(size_t)r * FBN + j] = s[j];
}

// pack graph weights [D][Fin][C] f32 -> WMMA B-fragment layout (bf16):
// idx = (((d*nch + kc)*8 + t)*32 + lane)*16 + j ; element = W[d][kc*32 + (lane/16)*8 + (j/8)*16 + j%8][t*16 + lane%16]
__global__ void pack_wg_kernel(const float* __restrict__ W, unsigned short* __restrict__ dst,
                               int Fin, int nch, int total) {
    int idx = blockIdx.x * 256 + threadIdx.x;
    if (idx >= total) return;
    int j    = idx & 15;
    int lane = (idx >> 4) & 31;
    int t    = (idx >> 9) & 7;
    int kc   = (idx >> 12) % nch;
    int d    = idx / (4096 * nch);
    int k = kc * 32 + (lane >> 4) * 8 + ((j & 8) << 1) + (j & 7);
    int n = t * 16 + (lane & 15);
    float v = (k < Fin) ? W[((size_t)d * Fin + k) * CC + n] : 0.f;
    dst[idx] = f2bf(v);
}

// pack Wconv [16][64][113] f32 -> WMMA A-fragment layout (bf16),
// padded K-space: k = n*128 + kk, kk in [0,128), zero for kk >= 113.
__global__ void pack_wc_kernel(const float* __restrict__ W, unsigned short* __restrict__ dst) {
    int idx = blockIdx.x * 256 + threadIdx.x;
    if (idx >= NCHC * 512) return;
    int j    = idx & 15;
    int lane = (idx >> 4) & 31;
    int kc   = idx >> 9;
    int m  = lane & 15;
    int hi = lane >> 4;
    int k  = kc * 32 + hi * 16 + j;   // A layout runs {0..15}/{16..31}
    int n  = k >> 7;
    int kk = k & 127;
    float v = (kk < KK) ? W[(size_t)m * (NN * KK) + n * KK + kk] : 0.f;
    dst[idx] = f2bf(v);
}

// =====================================================================
// fused graph-conv layer: gather(+BN/ReLU of prev) -> degree-masked GEMM
//                         -> bias -> z out (bf16) + BN stat reduction
// 256 threads, 8 waves, 128 rows per block.
// =====================================================================
__global__ __launch_bounds__(256)
void graph_layer_kernel(const float* __restrict__ inF32,            // layer1 atoms (or null)
                        const unsigned short* __restrict__ inBF,    // prev z bf16 (or null)
                        const float* __restrict__ scaleIn,
                        const float* __restrict__ shiftIn,
                        const int*   __restrict__ edges,
                        const float* __restrict__ bsum,
                        const unsigned short* __restrict__ wpack,
                        const float* __restrict__ bias,             // [5][128] f32
                        unsigned short* __restrict__ zout,          // [R][128] bf16
                        float* __restrict__ gsum, float* __restrict__ gsq,
                        int F, int nch)
{
    const int KPAD = nch * 32;
    __shared__ __align__(16) unsigned short sact[128 * 168];
    __shared__ int   sdeg[128];
    __shared__ float sbias[DD * CC];
    __shared__ float ssum[CC], ssq[CC];

    int tid = threadIdx.x;
    for (int i = tid; i < DD * CC; i += 256) sbias[i] = bias[i];
    if (tid < CC) { ssum[tid] = 0.f; ssq[tid] = 0.f; }

    // ---- gather + concat (2 threads per row) ----
    int rl = tid >> 1, half = tid & 1;
    int row = blockIdx.x * 128 + rl;
    int b = row >> 6;
    const int* ep = edges + (size_t)row * DD;
    int e[DD];
    int deg = 0;
    #pragma unroll
    for (int d = 0; d < DD; d++) { e[d] = ep[d]; deg += (e[d] >= 0); }
    if (!half) sdeg[rl] = deg;

    unsigned short* arow = &sact[rl * 168];
    if (inBF == nullptr) {                       // layer 1: raw atoms f32, F = 62
        const float* srcs[DD + 1]; int ns = 0;
        srcs[ns++] = inF32 + (size_t)row * FA;
        #pragma unroll
        for (int d = 0; d < DD; d++)
            if (e[d] >= 0) srcs[ns++] = inF32 + (size_t)(b * NN + e[d]) * FA;
        for (int f = half; f < F; f += 2) {
            float acc = 0.f;
            for (int s = 0; s < ns; s++) acc += srcs[s][f];
            arow[f] = f2bf(acc);
        }
    } else {                                     // layers 2/3: bf16 z + BN + ReLU on the fly
        const unsigned short* srcs[DD + 1]; int ns = 0;
        srcs[ns++] = inBF + (size_t)row * CC;
        #pragma unroll
        for (int d = 0; d < DD; d++)
            if (e[d] >= 0) srcs[ns++] = inBF + (size_t)(b * NN + e[d]) * CC;
        for (int f = half; f < F; f += 2) {
            float sc = scaleIn[f], sh = shiftIn[f];
            float acc = 0.f;
            for (int s = 0; s < ns; s++)
                acc += fmaxf(fmaf(bf2f(srcs[s][f]), sc, sh), 0.f);
            arow[f] = f2bf(acc);
        }
    }
    if (!half) {
        const float* bs = bsum + (size_t)row * FBN;
        #pragma unroll
        for (int j = 0; j < FBN; j++) arow[F + j] = f2bf(bs[j]);
    } else {
        for (int f = F + FBN; f < KPAD; f++) arow[f] = 0;   // zero K-pad
    }
    __syncthreads();

    // ---- degree-masked WMMA GEMM: one wave -> 16 rows x 128 cols ----
    int wave = tid >> 5, lane = tid & 31;
    int mrow = lane & 15, hi = lane >> 4;
    int lane_deg = sdeg[wave * 16 + mrow];

    v8f zerov = {0.f,0.f,0.f,0.f,0.f,0.f,0.f,0.f};
    v8f acc[8];
    #pragma unroll
    for (int t = 0; t < 8; t++) acc[t] = zerov;

    for (int kc = 0; kc < nch; kc++) {
        int k0 = kc * 32;
        Frag a;
        const unsigned short* ap = &sact[(wave * 16 + mrow) * 168 + k0 + hi * 8];
        a.q[0] = *(const uint4*)(ap);
        a.q[1] = *(const uint4*)(ap + 16);
        #pragma unroll
        for (int d = 0; d < DD; d++) {
            bool on = (lane_deg == d + 1);
            Frag am;
            #pragma unroll
            for (int i = 0; i < 8; i++) am.dw[i] = on ? a.dw[i] : 0u;
            const unsigned short* wp = wpack + (size_t)(d * nch + kc) * 4096 + (size_t)lane * 16;
            #pragma unroll
            for (int t = 0; t < 8; t++) {
                Frag w;
                const uint4* wq = (const uint4*)(wp + (size_t)t * 512);
                w.q[0] = wq[0]; w.q[1] = wq[1];
                acc[t] = __builtin_amdgcn_wmma_f32_16x16x32_bf16(
                    false, am.v, false, w.v, (short)0, acc[t], false, false);
            }
        }
    }

    // ---- bias, store z (bf16), LDS stat reduction ----
    int rowbase = blockIdx.x * 128 + wave * 16;
    #pragma unroll
    for (int t = 0; t < 8; t++) {
        int c = t * 16 + mrow;
        float psum = 0.f, psq = 0.f;
        #pragma unroll
        for (int v = 0; v < 8; v++) {
            int rloc = wave * 16 + v + 8 * hi;
            int dg = sdeg[rloc];
            float z = acc[t][v];
            z = (dg > 0) ? (z + sbias[(dg - 1) * CC + c]) : 0.f;
            zout[(size_t)(rowbase + v + 8 * hi) * CC + c] = f2bf(z);
            psum += z; psq += z * z;
        }
        atomicAdd(&ssum[c], psum);
        atomicAdd(&ssq[c],  psq);
    }
    __syncthreads();
    if (tid < CC) {
        atomicAdd(&gsum[tid], ssum[tid]);
        atomicAdd(&gsq[tid],  ssq[tid]);
    }
}

// =====================================================================
// BN finalize
// =====================================================================
__global__ void bn_finalize_kernel(const float* __restrict__ sum, const float* __restrict__ sq,
                                   const float* __restrict__ gamma, const float* __restrict__ beta,
                                   float* __restrict__ scale, float* __restrict__ shift,
                                   float invcnt, int n) {
    int i = threadIdx.x;
    if (i >= n) return;
    float m = sum[i] * invcnt;
    float var = sq[i] * invcnt - m * m;
    float inv = rsqrtf(var + EPS);
    float sc = gamma[i] * inv;
    scale[i] = sc;
    shift[i] = beta[i] - m * sc;
}

// =====================================================================
// Conv1d as WMMA GEMM with padded K (k = n*128 + kk):
// every 32-wide chunk lies inside one channel n, so the 16 B-fragment
// elements per lane are contiguous in LDS -> no division, 1 add per elem.
// 64 threads = 2 waves, one b per wave; LDS tile stride 144 (cols 128..143 zero).
// =====================================================================
#define CSTR 144
__global__ __launch_bounds__(64)
void conv_kernel(const unsigned short* __restrict__ z3,
                 const float* __restrict__ sc3, const float* __restrict__ sh3,
                 const unsigned short* __restrict__ wcpack,
                 float* __restrict__ z4, float* __restrict__ gsum, float* __restrict__ gsq)
{
    __shared__ __align__(16) unsigned short sx[2 * NN * CSTR];
    int tid = threadIdx.x, wave = tid >> 5, lane = tid & 31;
    int b = blockIdx.x * 2 + wave;
    unsigned short* slot = &sx[wave * NN * CSTR];

    const unsigned short* zp = z3 + (size_t)b * (NN * CC);
    for (int n = 0; n < NN; n++) {
        for (int c = lane; c < CSTR; c += 32) {
            float x = 0.f;
            if (c < CC) x = fmaxf(fmaf(bf2f(zp[n * CC + c]), sc3[c], sh3[c]), 0.f);
            slot[n * CSTR + c] = f2bf(x);
        }
    }
    __syncthreads();

    int l = lane & 15, hi = lane >> 4;
    v8f acc = {0.f,0.f,0.f,0.f,0.f,0.f,0.f,0.f};

    #pragma unroll 4
    for (int kc = 0; kc < NCHC; kc++) {
        Frag a;
        const uint4* ap = (const uint4*)(wcpack + ((size_t)kc * 32 + lane) * 16);
        a.q[0] = ap[0]; a.q[1] = ap[1];
        Frag bm;
        int kb   = kc * 32 + hi * 16;
        int n    = kb >> 7;
        int col0 = kb & 127;                        // 16-aligned, col0+15 <= 127
        const unsigned short* bp = &slot[n * CSTR + l + col0];
        #pragma unroll
        for (int j = 0; j < 16; j++) bm.u[j] = bp[j];
        acc = __builtin_amdgcn_wmma_f32_16x16x32_bf16(
            false, a.v, false, bm.v, (short)0, acc, false, false);
    }

    float psum = 0.f, psq = 0.f;
    #pragma unroll
    for (int v = 0; v < 8; v++) {
        int o = v + 8 * hi;
        float z = acc[v];
        z4[(size_t)b * 256 + o * 16 + l] = z;
        psum += z; psq += z * z;
    }
    atomicAdd(&gsum[l], psum);
    atomicAdd(&gsq[l],  psq);
}

__global__ void final_out_kernel(const float* __restrict__ z4,
                                 const float* __restrict__ sc4, const float* __restrict__ sh4,
                                 float* __restrict__ out) {
    int i = blockIdx.x * 256 + threadIdx.x;
    int l = i & 15;
    out[i] = fmaxf(fmaf(z4[i], sc4[l], sh4[l]), 0.f);
}

// =====================================================================
// host launcher
// =====================================================================
extern "C" void kernel_launch(void* const* d_in, const int* in_sizes, int n_in,
                              void* d_out, int out_size, void* d_ws, size_t ws_size,
                              hipStream_t stream)
{
    const float* atoms = (const float*)d_in[0];
    const float* bonds = (const float*)d_in[1];
    const int*   edges = (const int*)  d_in[2];
    const float* Wg1 = (const float*)d_in[3];  const float* bg1 = (const float*)d_in[4];
    const float* Wg2 = (const float*)d_in[5];  const float* bg2 = (const float*)d_in[6];
    const float* Wg3 = (const float*)d_in[7];  const float* bg3 = (const float*)d_in[8];
    const float* bn1_g = (const float*)d_in[9];  const float* bn1_b = (const float*)d_in[10];
    const float* bn2_g = (const float*)d_in[11]; const float* bn2_b = (const float*)d_in[12];
    const float* bn3_g = (const float*)d_in[13]; const float* bn3_b = (const float*)d_in[14];
    const float* Wconv = (const float*)d_in[15];
    const float* bn4_g = (const float*)d_in[16]; const float* bn4_b = (const float*)d_in[17];
    float* out = (float*)d_out;

    char*  ws  = (char*)d_ws;
    float* wsf = (float*)d_ws;
    float* bsum = (float*)(ws + OFF_BSUM);
    unsigned short* wp1 = (unsigned short*)(ws + OFF_WP1);
    unsigned short* wp2 = (unsigned short*)(ws + OFF_WP2);
    unsigned short* wp3 = (unsigned short*)(ws + OFF_WP3);
    unsigned short* wc  = (unsigned short*)(ws + OFF_WC);
    unsigned short* z1  = (unsigned short*)(ws + OFF_Z1);
    unsigned short* z2  = (unsigned short*)(ws + OFF_Z2);
    float* z4 = (float*)(ws + OFF_Z4);

    const float invR  = 1.0f / (float)RTOT;
    const float invBC = 1.0f / (float)(BB * COUT);

    // prep
    zero_stats_kernel<<<1, 1024, 0, stream>>>(wsf);
    bsum_kernel<<<RTOT / 256, 256, 0, stream>>>(bonds, bsum);
    pack_wg_kernel<<<(DD * 3 * 4096 + 255) / 256, 256, 0, stream>>>(Wg1, wp1, FA + FBN, 3, DD * 3 * 4096);
    pack_wg_kernel<<<(DD * 5 * 4096 + 255) / 256, 256, 0, stream>>>(Wg2, wp2, CC + FBN, 5, DD * 5 * 4096);
    pack_wg_kernel<<<(DD * 5 * 4096 + 255) / 256, 256, 0, stream>>>(Wg3, wp3, CC + FBN, 5, DD * 5 * 4096);
    pack_wc_kernel<<<(NCHC * 512 + 255) / 256, 256, 0, stream>>>(Wconv, wc);

    // layer 1: atoms -> z1
    graph_layer_kernel<<<RTOT / 128, 256, 0, stream>>>(
        atoms, nullptr, nullptr, nullptr, edges, bsum, wp1, bg1,
        z1, wsf + F_SUM1, wsf + F_SQ1, FA, 3);
    bn_finalize_kernel<<<1, 128, 0, stream>>>(wsf + F_SUM1, wsf + F_SQ1, bn1_g, bn1_b,
                                              wsf + F_SC1, wsf + F_SH1, invR, CC);
    // layer 2: z1 -> z2
    graph_layer_kernel<<<RTOT / 128, 256, 0, stream>>>(
        nullptr, z1, wsf + F_SC1, wsf + F_SH1, edges, bsum, wp2, bg2,
        z2, wsf + F_SUM2, wsf + F_SQ2, CC, 5);
    bn_finalize_kernel<<<1, 128, 0, stream>>>(wsf + F_SUM2, wsf + F_SQ2, bn2_g, bn2_b,
                                              wsf + F_SC2, wsf + F_SH2, invR, CC);
    // layer 3: z2 -> z1 (reuse)
    graph_layer_kernel<<<RTOT / 128, 256, 0, stream>>>(
        nullptr, z2, wsf + F_SC2, wsf + F_SH2, edges, bsum, wp3, bg3,
        z1, wsf + F_SUM3, wsf + F_SQ3, CC, 5);
    bn_finalize_kernel<<<1, 128, 0, stream>>>(wsf + F_SUM3, wsf + F_SQ3, bn3_g, bn3_b,
                                              wsf + F_SC3, wsf + F_SH3, invR, CC);
    // conv (applies BN3+ReLU while loading) -> z4 + BN4 stats
    conv_kernel<<<BB / 2, 64, 0, stream>>>(z1, wsf + F_SC3, wsf + F_SH3, wc,
                                           z4, wsf + F_SUM4, wsf + F_SQ4);
    bn_finalize_kernel<<<1, 16, 0, stream>>>(wsf + F_SUM4, wsf + F_SQ4, bn4_g, bn4_b,
                                             wsf + F_SC4, wsf + F_SH4, invBC, COUT);
    // BN4 + ReLU -> out [B,16,16]
    final_out_kernel<<<(BB * 256) / 256, 256, 0, stream>>>(z4, wsf + F_SC4, wsf + F_SH4, out);

    (void)in_sizes; (void)n_in; (void)out_size; (void)ws_size;
}
